// MACEModule_33303176413869
// MI455X (gfx1250) — compile-verified, matrix operand order checked by reading, not run on hardware
//
#include <hip/hip_runtime.h>
#include <hip/hip_bf16.h>

#define DEVFN __device__ __forceinline__

typedef __attribute__((ext_vector_type(16))) __bf16 bf16x16;
typedef __attribute__((ext_vector_type(8)))  float  f32x8;
typedef int v4i __attribute__((vector_size(16)));
typedef __attribute__((address_space(1))) v4i gv4i;   // global int4
typedef __attribute__((address_space(3))) v4i sv4i;   // LDS int4

constexpr int N     = 20000;
constexpr int E     = 320000;
constexpr int L     = 64;
constexpr int FEDGE = 136;
constexpr int K0PAD = 160;   // 136 padded up to 5*32
constexpr int OUTC  = 576;   // 64 * 9

// ---------------------------------------------------------------------------
// WMMA helpers (CDNA5 gfx1250, wave32, 16x16x32 bf16 -> f32)
// ---------------------------------------------------------------------------
DEVFN f32x8 wmma_bf16(bf16x16 a, bf16x16 b, f32x8 c) {
  // (neg_a, A, neg_b, B, c_mod, C, reuse_a, reuse_b)
  return __builtin_amdgcn_wmma_f32_16x16x32_bf16(false, a, false, b, (short)0, c,
                                                 false, false);
}

// 16-bit A-matrix 16x32 layout: lane half (0: lanes 0-15, 1: lanes 16-31)
// VGPR i<4 : K = 2i + 8*half + p ; VGPR i>=4 : K = 16 + 2(i-4) + 8*half + p
DEVFN int klocal(int i, int half) {
  return ((i >= 4) ? 16 : 0) + ((i & 3) << 1) + (half << 3);
}

// Build A fragment from one LDS row (row-major, bf16), K-block kb.
DEVFN bf16x16 load_a_row(const __bf16* row, int kb, int half) {
  bf16x16 a;
#pragma unroll
  for (int i = 0; i < 8; ++i) {
    int k = kb * 32 + klocal(i, half);
    a[2 * i]     = row[k];
    a[2 * i + 1] = row[k + 1];
  }
  return a;
}

// B fragments are pre-permuted in memory: tile*512 + lane*16 contiguous bf16.
DEVFN bf16x16 load_b_frag(const __bf16* frag, int tile, int lane) {
  return *(const bf16x16*)(frag + (size_t)tile * 512 + lane * 16);
}

DEVFN float silu(float x) { return x / (1.0f + __expf(-x)); }

// ---------------------------------------------------------------------------
// Zero message scratch and the zero-padded tail of sc output
// ---------------------------------------------------------------------------
__global__ void zero_kernel(float* __restrict__ msg, float* __restrict__ sc,
                            int count) {
  for (int i = blockIdx.x * blockDim.x + threadIdx.x; i < count;
       i += gridDim.x * blockDim.x) {
    msg[i] = 0.0f;
    sc[i]  = 0.0f;
  }
}

// ---------------------------------------------------------------------------
// Pre-permute a weight matrix (K x cols, row-major f32) into WMMA B-fragment
// order (bf16), folding the 1/sqrt(fan_in) scale. Zero-pads K -> Kpad.
// dst[gid] with gid = tile*512 + lane*16 + idx, tile = kb*(cols/16)+j
// ---------------------------------------------------------------------------
__global__ void frag_kernel(const float* __restrict__ src, __bf16* __restrict__ dst,
                            int K, int Kpad, int cols, float scale) {
  int jcount = cols >> 4;
  int total  = (Kpad >> 5) * jcount * 512;
  for (int gid = blockIdx.x * blockDim.x + threadIdx.x; gid < total;
       gid += gridDim.x * blockDim.x) {
    int t    = gid >> 9;
    int lane = (gid >> 4) & 31;
    int idx  = gid & 15;
    int kb   = t / jcount;
    int j    = t - kb * jcount;
    int n    = lane & 15, half = lane >> 4;
    int k    = kb * 32 + klocal(idx >> 1, half) + (idx & 1);
    float v  = (k < K) ? src[k * cols + j * 16 + n] * scale : 0.0f;
    dst[gid] = (__bf16)v;
  }
}

// ---------------------------------------------------------------------------
// Node kernel: x = node_feats @ W_up (scaled), and the skip connection
// sc[n,w] = sum_{u,e} x[n,u]*attrs[n,e]*W_skip[u,e,w] as a K=4096 GEMM with
// the rank-1 A matrix built on the fly. 1 block = 16 nodes, 4 waves = 4 jtiles.
// ---------------------------------------------------------------------------
__global__ void node_kernel(const float* __restrict__ node_feats,
                            const float* __restrict__ node_attrs,
                            const __bf16* __restrict__ wupf,
                            const __bf16* __restrict__ wskipf,
                            float* __restrict__ xbuf,
                            float* __restrict__ sc_out) {
  __shared__ __bf16 xa[16][72];
  __shared__ __bf16 na[16][72];
  __shared__ __bf16 xb[16][72];
  int t = threadIdx.x, lane = t & 31, w = t >> 5;
  int base = blockIdx.x * 16;

  for (int i = t; i < 16 * 64; i += 128) {
    int r = i >> 6, c = i & 63;
    xa[r][c] = (__bf16)node_feats[(base + r) * 64 + c];
    na[r][c] = (__bf16)node_attrs[(base + r) * 64 + c];
  }
  __syncthreads();

  int mrow = lane & 15, half = lane >> 4, n = lane & 15;

  // x tile (cols w*16 .. w*16+15)
  f32x8 acc;
#pragma unroll
  for (int i = 0; i < 8; ++i) acc[i] = 0.0f;
  for (int kb = 0; kb < 2; ++kb) {
    bf16x16 a = load_a_row(&xa[mrow][0], kb, half);
    bf16x16 b = load_b_frag(wupf, kb * 4 + w, lane);
    acc = wmma_bf16(a, b, acc);
  }
#pragma unroll
  for (int i = 0; i < 8; ++i) {
    int r = i + 8 * half, c = w * 16 + n;
    xbuf[(base + r) * 64 + c] = acc[i];
    xb[r][c]                  = (__bf16)acc[i];
  }
  __syncthreads();

  // skip: K = 4096 = 128 K-blocks, A'[m][k] = x[m][k>>6] * attrs[m][k&63]
  f32x8 s;
#pragma unroll
  for (int i = 0; i < 8; ++i) s[i] = 0.0f;
  for (int kb = 0; kb < 128; ++kb) {
    bf16x16 a;
#pragma unroll
    for (int i = 0; i < 8; ++i) {
      int k0 = kb * 32 + klocal(i, half);
#pragma unroll
      for (int p = 0; p < 2; ++p) {
        int k = k0 + p;
        int u = k >> 6, e = k & 63;
        a[2 * i + p] = (__bf16)((float)xb[mrow][u] * (float)na[mrow][e]);
      }
    }
    bf16x16 b = load_b_frag(wskipf, kb * 4 + w, lane);
    s = wmma_bf16(a, b, s);
  }
#pragma unroll
  for (int i = 0; i < 8; ++i) {
    int r = i + 8 * half, c = w * 16 + n;
    sc_out[(base + r) * OUTC + c] = s[i];
  }
}

// ---------------------------------------------------------------------------
// One MLP stage: (16 x K) @ (K x 64) GEMM + SiLU, LDS -> LDS, per wave.
// ---------------------------------------------------------------------------
DEVFN void mlp_stage(const __bf16* src, int sstride, const __bf16* wfrag,
                     int nkb, __bf16* dst, int dstride, int lane) {
  int mrow = lane & 15, half = lane >> 4, n = lane & 15;
  for (int j = 0; j < 4; ++j) {
    f32x8 acc;
#pragma unroll
    for (int i = 0; i < 8; ++i) acc[i] = 0.0f;
    for (int kb = 0; kb < nkb; ++kb) {
      bf16x16 a = load_a_row(src + mrow * sstride, kb, half);
      bf16x16 b = load_b_frag(wfrag, kb * 4 + j, lane);
      acc = wmma_bf16(a, b, acc);
    }
#pragma unroll
    for (int i = 0; i < 8; ++i) {
      int r = i + 8 * half;
      dst[r * dstride + j * 16 + n] = (__bf16)silu(acc[i]);
    }
  }
}

// ---------------------------------------------------------------------------
// Edge kernel: 64 edges/block, 16 edges/wave. 4-layer WMMA MLP in LDS, then
// gate by x[sender] and scatter-add spherical-harmonic products into msg.
// x[sender] rows are gathered with GLOBAL_LOAD_ASYNC_TO_LDS_B128 (ASYNCcnt)
// so the gather overlaps the bf16 staging of edge_feats.
// ---------------------------------------------------------------------------
__global__ void edge_kernel(const float* __restrict__ edge_feats,
                            const float* __restrict__ edge_attrs,
                            const int*   __restrict__ edge_index,
                            const float* __restrict__ xbuf,
                            const __bf16* __restrict__ w0f,
                            const __bf16* __restrict__ w1f,
                            const __bf16* __restrict__ w2f,
                            const __bf16* __restrict__ w3f,
                            float* __restrict__ msg) {
  __shared__ __bf16 ef[4][16][K0PAD + 8];
  __shared__ __bf16 ha[4][16][72];
  __shared__ __bf16 hb[4][16][72];
  __shared__ float  xs[4][16][68];   // row stride 68*4 = 272 B (16B aligned)
  __shared__ float  ea[4][16][12];
  __shared__ int    rcv[4][16];

  int t = threadIdx.x, lane = t & 31, w = t >> 5;
  int myBase = blockIdx.x * 64 + w * 16;

  // warm L2 with the layer-3 weight fragments used at the end of the pipeline
  __builtin_prefetch(w3f + (size_t)lane * 64, 0, 1);

  // async gather of x[sender] rows straight into LDS (no VGPR staging):
  // 16 rows x 16 chunks of 16 B each, per wave.
  for (int i = lane; i < 16 * 16; i += 32) {
    int r = i >> 4, c4 = i & 15;
    int snd = edge_index[myBase + r];
    const float* gsrc = xbuf + (size_t)snd * 64 + c4 * 4;
    __builtin_amdgcn_global_load_async_to_lds_b128(
        (gv4i*)gsrc, (sv4i*)&xs[w][r][c4 * 4], 0, 0);
  }

  // stage edge features (f32 -> bf16, zero-padded K) while the gather flies
  for (int i = lane; i < 16 * K0PAD; i += 32) {
    int r = i / K0PAD, c = i - r * K0PAD;
    float v = (c < FEDGE) ? edge_feats[(size_t)(myBase + r) * FEDGE + c] : 0.0f;
    ef[w][r][c] = (__bf16)v;
  }
  for (int i = lane; i < 16 * 9; i += 32) {
    int r = i / 9, c = i - r * 9;
    ea[w][r][c] = edge_attrs[(size_t)(myBase + r) * 9 + c];
  }
  if (lane < 16) rcv[w][lane] = edge_index[E + myBase + lane];

  __builtin_amdgcn_s_wait_asynccnt(0);   // wave-local: xs is produced/consumed
  __syncthreads();                        // by the same wave's lanes

  mlp_stage(&ef[w][0][0], K0PAD + 8, w0f, 5, &ha[w][0][0], 72, lane);
  __syncthreads();
  mlp_stage(&ha[w][0][0], 72, w1f, 2, &hb[w][0][0], 72, lane);
  __syncthreads();
  mlp_stage(&hb[w][0][0], 72, w2f, 2, &ha[w][0][0], 72, lane);
  __syncthreads();

  // layer 3: 192 output cols = 12 j-tiles; gate + scatter immediately
  int mrow = lane & 15, half = lane >> 4, n = lane & 15;
  for (int j3 = 0; j3 < 12; ++j3) {
    f32x8 acc;
#pragma unroll
    for (int i = 0; i < 8; ++i) acc[i] = 0.0f;
    for (int kb = 0; kb < 2; ++kb) {
      bf16x16 a = load_a_row(&ha[w][mrow][0], kb, half);
      bf16x16 b = load_b_frag(w3f, kb * 12 + j3, lane);
      acc = wmma_bf16(a, b, acc);
    }
    int c = j3 * 16 + n;        // 0..191
    int g = c >> 6;             // channel 0/1/2
    int u = c & 63;
    int mlo = (g == 0) ? 0 : ((g == 1) ? 1 : 4);
    int mhi = (g == 0) ? 1 : ((g == 1) ? 4 : 9);
#pragma unroll
    for (int i = 0; i < 8; ++i) {
      int r = i + 8 * half;
      float gv = acc[i] * xs[w][r][u];
      float* dst = msg + (size_t)rcv[w][r] * OUTC + u * 9;
      for (int mm = mlo; mm < mhi; ++mm)
        atomicAdd(dst + mm, gv * ea[w][r][mm]);
    }
  }
}

// ---------------------------------------------------------------------------
// Out kernel: per m-channel (9), out[:,:,m] = msg[:,:,m] @ W_lin{g} * inv
// (inv folded into fragments). 16 nodes/block, 4 waves = 4 jtiles.
// ---------------------------------------------------------------------------
__global__ void out_kernel(const float* __restrict__ msg,
                           const __bf16* __restrict__ lin0f,
                           const __bf16* __restrict__ lin1f,
                           const __bf16* __restrict__ lin2f,
                           float* __restrict__ outp) {
  __shared__ __bf16 mt[16][72];
  int t = threadIdx.x, lane = t & 31, w = t >> 5;
  int base = blockIdx.x * 16;
  int mrow = lane & 15, half = lane >> 4, n = lane & 15;

  for (int m = 0; m < 9; ++m) {
    __syncthreads();
    for (int i = t; i < 16 * 64; i += 128) {
      int r = i >> 6, c = i & 63;
      mt[r][c] = (__bf16)msg[(size_t)(base + r) * OUTC + c * 9 + m];
    }
    __syncthreads();
    const __bf16* lf = (m == 0) ? lin0f : ((m < 4) ? lin1f : lin2f);
    f32x8 acc;
#pragma unroll
    for (int i = 0; i < 8; ++i) acc[i] = 0.0f;
    for (int kb = 0; kb < 2; ++kb) {
      bf16x16 a = load_a_row(&mt[mrow][0], kb, half);
      bf16x16 b = load_b_frag(lf, kb * 4 + w, lane);
      acc = wmma_bf16(a, b, acc);
    }
#pragma unroll
    for (int i = 0; i < 8; ++i) {
      int r = i + 8 * half, v = w * 16 + n;
      outp[((size_t)(base + r) * 64 + v) * 9 + m] = acc[i];
    }
  }
}

// ---------------------------------------------------------------------------
// Host launcher
// ---------------------------------------------------------------------------
extern "C" void kernel_launch(void* const* d_in, const int* in_sizes, int n_in,
                              void* d_out, int out_size, void* d_ws, size_t ws_size,
                              hipStream_t stream) {
  (void)in_sizes; (void)n_in; (void)out_size; (void)ws_size;

  const float* node_attrs = (const float*)d_in[0];
  const float* node_feats = (const float*)d_in[1];
  const float* edge_attrs = (const float*)d_in[2];
  const float* edge_feats = (const float*)d_in[3];
  const float* W_up       = (const float*)d_in[4];
  const float* mlp_w0     = (const float*)d_in[5];
  const float* mlp_w1     = (const float*)d_in[6];
  const float* mlp_w2     = (const float*)d_in[7];
  const float* mlp_w3     = (const float*)d_in[8];
  const float* W_lin0     = (const float*)d_in[9];
  const float* W_lin1     = (const float*)d_in[10];
  const float* W_lin2     = (const float*)d_in[11];
  const float* W_skip     = (const float*)d_in[12];
  const int*   edge_index = (const int*)d_in[13];

  float* outp = (float*)d_out;
  float* sc   = outp + (size_t)N * OUTC;

  // workspace layout
  char* ws = (char*)d_ws;
  constexpr size_t MSG_BYTES  = (size_t)N * OUTC * 4;           // 46.08 MB
  constexpr size_t XBUF_BYTES = (size_t)N * L * 4;              // 5.12 MB
  float*  msg      = (float*)ws;
  float*  xbuf     = (float*)(ws + MSG_BYTES);
  __bf16* fragbase = (__bf16*)(ws + MSG_BYTES + XBUF_BYTES);
  // fragment sub-buffers (elements)
  constexpr size_t F_WUP   = 0;
  constexpr size_t F_W0    = F_WUP  + 8  * 512;
  constexpr size_t F_W1    = F_W0   + 20 * 512;
  constexpr size_t F_W2    = F_W1   + 8  * 512;
  constexpr size_t F_W3    = F_W2   + 8  * 512;
  constexpr size_t F_LIN0  = F_W3   + 24 * 512;
  constexpr size_t F_LIN1  = F_LIN0 + 8  * 512;
  constexpr size_t F_LIN2  = F_LIN1 + 8  * 512;
  constexpr size_t F_WSKIP = F_LIN2 + 8  * 512;
  __bf16* wupf   = fragbase + F_WUP;
  __bf16* w0f    = fragbase + F_W0;
  __bf16* w1f    = fragbase + F_W1;
  __bf16* w2f    = fragbase + F_W2;
  __bf16* w3f    = fragbase + F_W3;
  __bf16* lin0f  = fragbase + F_LIN0;
  __bf16* lin1f  = fragbase + F_LIN1;
  __bf16* lin2f  = fragbase + F_LIN2;
  __bf16* wskipf = fragbase + F_WSKIP;

  const float s8    = 0.125f;               // 1/sqrt(64)
  const float s136  = 0.085749292571254f;   // 1/sqrt(136)
  const float sLin  = 0.0125f;              // (1/AVG_NEIGH)/sqrt(64)
  const float sSkip = 0.015625f;            // 1/sqrt(64*64)

  zero_kernel<<<2048, 256, 0, stream>>>(msg, sc, N * OUTC);

  frag_kernel<<<512, 256, 0, stream>>>(W_up,   wupf,   64,   64,   64,  s8);
  frag_kernel<<<512, 256, 0, stream>>>(mlp_w0, w0f,    136,  K0PAD,64,  s136);
  frag_kernel<<<512, 256, 0, stream>>>(mlp_w1, w1f,    64,   64,   64,  s8);
  frag_kernel<<<512, 256, 0, stream>>>(mlp_w2, w2f,    64,   64,   64,  s8);
  frag_kernel<<<512, 256, 0, stream>>>(mlp_w3, w3f,    64,   64,   192, s8);
  frag_kernel<<<512, 256, 0, stream>>>(W_lin0, lin0f,  64,   64,   64,  sLin);
  frag_kernel<<<512, 256, 0, stream>>>(W_lin1, lin1f,  64,   64,   64,  sLin);
  frag_kernel<<<512, 256, 0, stream>>>(W_lin2, lin2f,  64,   64,   64,  sLin);
  frag_kernel<<<1024, 256, 0, stream>>>(W_skip, wskipf, 4096, 4096, 64, sSkip);

  node_kernel<<<N / 16, 128, 0, stream>>>(node_feats, node_attrs, wupf, wskipf,
                                          xbuf, sc);
  edge_kernel<<<E / 64, 128, 0, stream>>>(edge_feats, edge_attrs, edge_index,
                                          xbuf, w0f, w1f, w2f, w3f, msg);
  out_kernel<<<N / 16, 128, 0, stream>>>(msg, lin0f, lin1f, lin2f, outp);
}
